// ProxPC_16063177687682
// MI455X (gfx1250) — matrix operand order repaired and checked by reading, not compiled, not run on hardware
//
#include <hip/hip_runtime.h>

// ---------------------------------------------------------------------------
// ProxPC predictive-coding network on MI455X (gfx1250, wave32, WMMA).
// Weights pre-converted (and pre-transposed for backward) to bf16 once per
// launch; GEMMs use v_wmma_f32_16x16x32_bf16 with f32 accumulate, B-tiles
// moved global->LDS via async-to-LDS DMA (global_load_async_to_lds_b128).
// ---------------------------------------------------------------------------

#define LRATE  0.001f
#define GAMMA  0.02f
#define NITER  25

typedef __bf16 bf16_t;
typedef __attribute__((ext_vector_type(16))) __bf16 v16bf;
typedef __attribute__((ext_vector_type(8)))  __bf16 v8bf;
typedef __attribute__((ext_vector_type(4)))  __bf16 v4bf;
typedef __attribute__((ext_vector_type(8)))  float  v8f;
typedef int v4i_vs __attribute__((vector_size(16)));   // matches builtin param pointee

#define BM 128
#define BN 128
#define BKK 32
#define LDK 40   // padded LDS row stride in bf16 elems (80 B, 16B-aligned rows)

#if defined(__has_builtin)
#if __has_builtin(__builtin_amdgcn_global_load_async_to_lds_b128)
#define HAVE_ASYNC_LDS 1
#endif
#endif

__device__ __forceinline__ bf16_t f2bf(float f) {
  union { float f; unsigned u; } v; v.f = f;
  unsigned r = v.u + 0x7FFFu + ((v.u >> 16) & 1u);   // round-to-nearest-even
  unsigned short h = (unsigned short)(r >> 16);
  return __builtin_bit_cast(bf16_t, h);
}

__device__ __forceinline__ void wait_async0() {
#if defined(__has_builtin) && __has_builtin(__builtin_amdgcn_s_wait_asynccnt)
  __builtin_amdgcn_s_wait_asynccnt(0);
#else
  asm volatile("s_wait_asynccnt 0x0" ::: "memory");
#endif
}

// C[M,N] = op(A) * Wb (+ bias)
//   AMODE 1: op(A) = relu(A1)     AMODE 2: op(A) = A1 - A2
//   Wb: bf16, row-major [N][K] (row n holds the K-vector multiplying into col n)
template<int AMODE, int HASBIAS>
__global__ __launch_bounds__(256) void gemm_bf16_wmma(
    const float* __restrict__ A1, const float* __restrict__ A2,
    const bf16_t* __restrict__ Wb, const float* __restrict__ bias,
    float* __restrict__ C, int M, int N, int K)
{
  __shared__ __align__(16) bf16_t As[BM * LDK];
  __shared__ __align__(16) bf16_t Bs[BN * LDK];

  const int tid  = threadIdx.x;
  const int wave = tid >> 5;
  const int lane = tid & 31;
  const int l16  = lane & 15;
  const int hsel = lane >> 4;          // half-wave select (0: lanes 0-15)
  const int bm   = blockIdx.y * BM;
  const int bn   = blockIdx.x * BN;
  const int wm   = (wave & 1) * 64;    // 2 waves along M
  const int wn   = (wave >> 1) * 32;   // 4 waves along N

  v8f acc[4][2];
  #pragma unroll
  for (int i = 0; i < 4; ++i)
    #pragma unroll
    for (int j = 0; j < 2; ++j)
      #pragma unroll
      for (int e = 0; e < 8; ++e) acc[i][j][e] = 0.f;

  const int lr = tid >> 3;   // 0..31 : A row within a 32-row pass
  const int lc = tid & 7;    // A float4 slot within 32-wide K tile
  const int br = tid >> 2;   // 0..63 : B row
  const int bs = tid & 3;    // B 16-byte segment within a 64B row

  for (int k0 = 0; k0 < K; k0 += BKK) {
    // ---- B tile: pure bf16 byte move, global -> LDS (async DMA) -------
#if defined(HAVE_ASYNC_LDS)
    {
      const bf16_t* g0 = Wb + (size_t)(bn + br) * K + k0 + bs * 8;
      const bf16_t* g1 = Wb + (size_t)(bn + br + 64) * K + k0 + bs * 8;
      __builtin_amdgcn_global_load_async_to_lds_b128(
          (__attribute__((address_space(1))) v4i_vs*)(const_cast<bf16_t*>(g0)),
          (__attribute__((address_space(3))) v4i_vs*)(Bs + br * LDK + bs * 8), 0, 0);
      __builtin_amdgcn_global_load_async_to_lds_b128(
          (__attribute__((address_space(1))) v4i_vs*)(const_cast<bf16_t*>(g1)),
          (__attribute__((address_space(3))) v4i_vs*)(Bs + (br + 64) * LDK + bs * 8), 0, 0);
    }
#else
    #pragma unroll
    for (int p = 0; p < 2; ++p) {
      const int r = br + p * 64;
      const v8bf v = *(const v8bf*)(Wb + (size_t)(bn + r) * K + k0 + bs * 8);
      *(v8bf*)(Bs + r * LDK + bs * 8) = v;
    }
#endif
    // ---- A tile: fp32 -> bf16 with fused relu / diff (overlaps B DMA) --
    #pragma unroll
    for (int p = 0; p < 4; ++p) {
      const int r = lr + p * 32;
      float4 v = *(const float4*)(A1 + (size_t)(bm + r) * K + k0 + lc * 4);
      if (AMODE == 1) {
        v.x = fmaxf(v.x, 0.f); v.y = fmaxf(v.y, 0.f);
        v.z = fmaxf(v.z, 0.f); v.w = fmaxf(v.w, 0.f);
      } else if (AMODE == 2) {
        const float4 u = *(const float4*)(A2 + (size_t)(bm + r) * K + k0 + lc * 4);
        v.x -= u.x; v.y -= u.y; v.z -= u.z; v.w -= u.w;
      }
      v4bf pk; pk[0] = f2bf(v.x); pk[1] = f2bf(v.y); pk[2] = f2bf(v.z); pk[3] = f2bf(v.w);
      *(v4bf*)(As + r * LDK + lc * 4) = pk;
    }
#if defined(HAVE_ASYNC_LDS)
    wait_async0();
#endif
    __syncthreads();

    // ---- fragments per ISA 16-bit layouts ----
    v16bf af[4];
    #pragma unroll
    for (int i = 0; i < 4; ++i) {
      const bf16_t* rp = As + (wm + i * 16 + l16) * LDK + hsel * 8;
      const v8bf lo = *(const v8bf*)(rp);        // K = kb .. kb+7
      const v8bf hi = *(const v8bf*)(rp + 16);   // K = kb+16 .. kb+23
      af[i] = __builtin_shufflevector(lo, hi, 0,1,2,3,4,5,6,7,8,9,10,11,12,13,14,15);
    }
    v16bf bfv[2];
    #pragma unroll
    for (int j = 0; j < 2; ++j) {
      const bf16_t* bp = Bs + (wn + j * 16 + l16) * LDK + hsel * 8;
      const v8bf blo = *(const v8bf*)(bp);
      const v8bf bhi = *(const v8bf*)(bp + 16);
      bfv[j] = __builtin_shufflevector(blo, bhi, 0,1,2,3,4,5,6,7,8,9,10,11,12,13,14,15);
    }

    #pragma unroll
    for (int i = 0; i < 4; ++i)
      #pragma unroll
      for (int j = 0; j < 2; ++j)
        acc[i][j] = __builtin_amdgcn_wmma_f32_16x16x32_bf16(
            false, af[i], false, bfv[j], (short)0, acc[i][j], false, false);
    __syncthreads();
  }

  // ---- epilogue: 16x16 f32 D layout (lane<16: row e, lane>=16: row e+8) ----
  #pragma unroll
  for (int i = 0; i < 4; ++i) {
    #pragma unroll
    for (int j = 0; j < 2; ++j) {
      const int row0 = bm + wm + i * 16 + hsel * 8;
      const int col  = bn + wn + j * 16 + l16;
      const float bv = HASBIAS ? bias[col] : 0.f;
      #pragma unroll
      for (int e = 0; e < 8; ++e)
        C[(size_t)(row0 + e) * N + col] = acc[i][j][e] + bv;
    }
  }
}

// ---------------------------------------------------------------------------
// One-time (per launch) weight preparation: fp32 -> bf16, plus transposes.
// ---------------------------------------------------------------------------
__global__ void convert_bf16_k(const float* __restrict__ src, bf16_t* __restrict__ dst, int n4) {
  const int i = blockIdx.x * blockDim.x + threadIdx.x;
  if (i < n4) {
    const float4 v = *(const float4*)(src + (size_t)i * 4);
    v4bf pk; pk[0] = f2bf(v.x); pk[1] = f2bf(v.y); pk[2] = f2bf(v.z); pk[3] = f2bf(v.w);
    *(v4bf*)(dst + (size_t)i * 4) = pk;
  }
}

// dst[c*R + r] = bf16(src[r*C + c]) ; src is [R][C] fp32
__global__ void transpose_bf16_k(const float* __restrict__ src, bf16_t* __restrict__ dst,
                                 int R, int Ccols) {
  __shared__ float tile[32][33];
  const int c0 = blockIdx.x * 32, r0 = blockIdx.y * 32;
  const int tx = threadIdx.x & 31, ty = threadIdx.x >> 5;   // 256 thr: ty 0..7
  #pragma unroll
  for (int i = 0; i < 32; i += 8)
    tile[ty + i][tx] = src[(size_t)(r0 + ty + i) * Ccols + c0 + tx];
  __syncthreads();
  #pragma unroll
  for (int i = 0; i < 32; i += 8)
    dst[(size_t)(c0 + ty + i) * R + r0 + tx] = f2bf(tile[tx][ty + i]);
}

// ---------------------------------------------------------------------------
// Elementwise / reduction kernels
// ---------------------------------------------------------------------------
__global__ void zero1_k(float* p) { if (blockIdx.x == 0 && threadIdx.x == 0) p[0] = 0.f; }

__global__ void relu_sq_sum_k(const float* __restrict__ X, int n, float* out) {
  __shared__ float red[256];
  float s = 0.f;
  for (int i = blockIdx.x * blockDim.x + threadIdx.x; i < n; i += gridDim.x * blockDim.x) {
    const float v = fmaxf(X[i], 0.f); s += v * v;
  }
  red[threadIdx.x] = s; __syncthreads();
  for (int st = 128; st > 0; st >>= 1) {
    if (threadIdx.x < st) red[threadIdx.x] += red[threadIdx.x + st];
    __syncthreads();
  }
  if (threadIdx.x == 0) atomicAdd(out, red[0]);
}

__global__ void softmax_rows_k(const float* __restrict__ X, float* __restrict__ Y, int N) {
  const int row = blockIdx.x;
  const float* x = X + (size_t)row * N;
  float* y = Y + (size_t)row * N;
  __shared__ float red[256];
  float m = -3.4e38f;
  for (int i = threadIdx.x; i < N; i += blockDim.x) m = fmaxf(m, x[i]);
  red[threadIdx.x] = m; __syncthreads();
  for (int st = 128; st > 0; st >>= 1) {
    if (threadIdx.x < st) red[threadIdx.x] = fmaxf(red[threadIdx.x], red[threadIdx.x + st]);
    __syncthreads();
  }
  m = red[0]; __syncthreads();
  float s = 0.f;
  for (int i = threadIdx.x; i < N; i += blockDim.x) s += __expf(x[i] - m);
  red[threadIdx.x] = s; __syncthreads();
  for (int st = 128; st > 0; st >>= 1) {
    if (threadIdx.x < st) red[threadIdx.x] += red[threadIdx.x + st];
    __syncthreads();
  }
  const float inv = 1.f / red[0];
  for (int i = threadIdx.x; i < N; i += blockDim.x) y[i] = __expf(x[i] - m) * inv;
}

// t = t - GAMMA*(t - p) + GAMMA*g*(t > 0)
__global__ void update_t_k(float* __restrict__ t, const float* __restrict__ p,
                           const float* __restrict__ g, int n) {
  const int i = blockIdx.x * blockDim.x + threadIdx.x;
  if (i < n) {
    const float tv = t[i];
    t[i] = tv - GAMMA * (tv - p[i]) + GAMMA * g[i] * (tv > 0.f ? 1.f : 0.f);
  }
}

// same + accumulate sum(relu(t_new)^2) into nmsum
__global__ void update_t_nm_k(float* __restrict__ t, const float* __restrict__ p,
                              const float* __restrict__ g, int n, float* nmsum) {
  __shared__ float red[256];
  float s = 0.f;
  const int i = blockIdx.x * blockDim.x + threadIdx.x;
  if (i < n) {
    const float tv = t[i];
    const float tn = tv - GAMMA * (tv - p[i]) + GAMMA * g[i] * (tv > 0.f ? 1.f : 0.f);
    t[i] = tn;
    const float r = fmaxf(tn, 0.f); s = r * r;
  }
  red[threadIdx.x] = s; __syncthreads();
  for (int st = 128; st > 0; st >>= 1) {
    if (threadIdx.x < st) red[threadIdx.x] += red[threadIdx.x + st];
    __syncthreads();
  }
  if (threadIdx.x == 0) atomicAdd(nmsum, red[0]);
}

// out = a*sm + (1-a)*tgt,  a = 1/(1 + (nmsum*scale + cbias)*LR)
__global__ void mix_t4_k(const float* __restrict__ sm, const float* __restrict__ tgt,
                         const float* __restrict__ nmsum, float scale, float cbias,
                         float* __restrict__ out, int n) {
  const float nm = nmsum[0] * scale + cbias;
  const float a  = 1.f / (1.f + nm * LRATE);
  const int i = blockIdx.x * blockDim.x + threadIdx.x;
  if (i < n) out[i] = a * sm[i] + (1.f - a) * tgt[i];
}

// ---------------------------------------------------------------------------
// Host orchestration
// ---------------------------------------------------------------------------
extern "C" void kernel_launch(void* const* d_in, const int* in_sizes, int n_in,
                              void* d_out, int out_size, void* d_ws, size_t ws_size,
                              hipStream_t stream) {
  (void)in_sizes; (void)n_in; (void)out_size; (void)ws_size;
  const int Bsz = 1024, D = 2048, O = 1024;

  const float* x   = (const float*)d_in[0];
  const float* tgt = (const float*)d_in[1];
  const float* W0  = (const float*)d_in[2]; const float* b0 = (const float*)d_in[3];
  const float* W1  = (const float*)d_in[4]; const float* b1 = (const float*)d_in[5];
  const float* W2  = (const float*)d_in[6]; const float* b2 = (const float*)d_in[7];
  const float* W3  = (const float*)d_in[8]; const float* b3 = (const float*)d_in[9];

  const size_t S2 = (size_t)Bsz * D, S1 = (size_t)Bsz * O;
  const size_t WDD = (size_t)D * D, WOD = (size_t)O * D;

  float* ws  = (float*)d_ws;
  float* t1  = ws;           float* p1 = ws + 1 * S2;
  float* t2  = ws + 2 * S2;  float* p2 = ws + 3 * S2;
  float* t3  = ws + 4 * S2;  float* p3 = ws + 5 * S2;
  float* g   = ws + 6 * S2;
  float* t4  = ws + 7 * S2;
  float* p4  = t4 + S1;
  float* sm4 = t4 + 2 * S1;
  float* nm  = t4 + 3 * S1;

  // bf16 weight area (16B aligned: all float offsets above are multiples of 1M)
  bf16_t* bw  = (bf16_t*)(nm + 16);
  bf16_t* W0b = bw;             // [D][D]  forward: @W0^T
  bf16_t* W1b = W0b + WDD;      // [D][D]  forward: @W1^T
  bf16_t* W2b = W1b + WDD;      // [D][D]  forward: @W2^T
  bf16_t* W3b = W2b + WDD;      // [O][D]  forward: @W3^T
  bf16_t* W1t = W3b + WOD;      // [D][D]  backward: e2 @ W1
  bf16_t* W2t = W1t + WDD;      // [D][D]  backward: e3 @ W2
  bf16_t* W3t = W2t + WDD;      // [D][O]  backward: e4 @ W3

  const int n2 = (int)S2, n1 = (int)S1;
  const dim3 blk(256);
  const dim3 gDD(D / BN, Bsz / BM);   // N=2048 outputs
  const dim3 gDO(O / BN, Bsz / BM);   // N=1024 outputs
  const int e2 = (n2 + 255) / 256, e1 = (n1 + 255) / 256;

  // ---- per-launch weight prep (bf16 + transposes) ----
  convert_bf16_k<<<(int)(WDD / 4 / 256), blk, 0, stream>>>(W0, W0b, (int)(WDD / 4));
  convert_bf16_k<<<(int)(WDD / 4 / 256), blk, 0, stream>>>(W1, W1b, (int)(WDD / 4));
  convert_bf16_k<<<(int)(WDD / 4 / 256), blk, 0, stream>>>(W2, W2b, (int)(WDD / 4));
  convert_bf16_k<<<(int)(WOD / 4 / 256), blk, 0, stream>>>(W3, W3b, (int)(WOD / 4));
  transpose_bf16_k<<<dim3(D / 32, D / 32), blk, 0, stream>>>(W1, W1t, D, D);
  transpose_bf16_k<<<dim3(D / 32, D / 32), blk, 0, stream>>>(W2, W2t, D, D);
  transpose_bf16_k<<<dim3(D / 32, O / 32), blk, 0, stream>>>(W3, W3t, O, D);

  // ---- feedforward init: t_i = h_i, p4 = h4 ----
  gemm_bf16_wmma<1,1><<<gDD, blk, 0, stream>>>(x,  nullptr, W0b, b0, t1, Bsz, D, D);
  gemm_bf16_wmma<1,1><<<gDD, blk, 0, stream>>>(t1, nullptr, W1b, b1, t2, Bsz, D, D);
  gemm_bf16_wmma<1,1><<<gDD, blk, 0, stream>>>(t2, nullptr, W2b, b2, t3, Bsz, D, D);
  gemm_bf16_wmma<1,1><<<gDO, blk, 0, stream>>>(t3, nullptr, W3b, b3, p4, Bsz, O, D);
  (void)hipMemcpyAsync(p1, t1, S2 * sizeof(float), hipMemcpyDeviceToDevice, stream);
  (void)hipMemcpyAsync(p2, t2, S2 * sizeof(float), hipMemcpyDeviceToDevice, stream);
  (void)hipMemcpyAsync(p3, t3, S2 * sizeof(float), hipMemcpyDeviceToDevice, stream);

  // nm0 = mean(sum(relu(h3)^2, axis=1)) + EPS ;  t4 = a0*softmax(h4) + (1-a0)*tgt
  zero1_k<<<1, 1, 0, stream>>>(nm);
  relu_sq_sum_k<<<512, blk, 0, stream>>>(t3, n2, nm);
  softmax_rows_k<<<Bsz, blk, 0, stream>>>(p4, sm4, O);
  mix_t4_k<<<e1, blk, 0, stream>>>(sm4, tgt, nm, 1.f / (float)Bsz, 1.0f /*EPS*/, t4, n1);

  // ---- 25 PC iterations ----
  for (int it = 0; it < NITER; ++it) {
    softmax_rows_k<<<Bsz, blk, 0, stream>>>(p4, sm4, O);               // sm4 = softmax(p4)

    gemm_bf16_wmma<2,0><<<gDD, blk, 0, stream>>>(t2, p2, W1t, nullptr, g, Bsz, D, D);  // g = e2 @ W1
    update_t_k<<<e2, blk, 0, stream>>>(t1, p1, g, n2);

    gemm_bf16_wmma<2,0><<<gDD, blk, 0, stream>>>(t3, p3, W2t, nullptr, g, Bsz, D, D);  // g = e3 @ W2
    update_t_k<<<e2, blk, 0, stream>>>(t2, p2, g, n2);

    gemm_bf16_wmma<2,0><<<gDD, blk, 0, stream>>>(t4, sm4, W3t, nullptr, g, Bsz, D, O); // g = e4 @ W3
    zero1_k<<<1, 1, 0, stream>>>(nm);
    update_t_nm_k<<<e2, blk, 0, stream>>>(t3, p3, g, n2, nm);          // t3 update + sum(relu^2)

    mix_t4_k<<<e1, blk, 0, stream>>>(sm4, tgt, nm, 1.f, 2.0f /*BIAS_C+EPS*/, t4, n1);

    gemm_bf16_wmma<1,1><<<gDD, blk, 0, stream>>>(t1, nullptr, W1b, b1, p2, Bsz, D, D); // p refresh
    gemm_bf16_wmma<1,1><<<gDD, blk, 0, stream>>>(t2, nullptr, W2b, b2, p3, Bsz, D, D);
    gemm_bf16_wmma<1,1><<<gDO, blk, 0, stream>>>(t3, nullptr, W3b, b3, p4, Bsz, O, D);
  }

  (void)hipMemcpyAsync(d_out, t4, S1 * sizeof(float), hipMemcpyDeviceToDevice, stream);
}
